// SplitAttention_76081050681597
// MI455X (gfx1250) — matrix operand order repaired and checked by reading, not compiled, not run on hardware
//
#include <hip/hip_runtime.h>
#include <hip/hip_bf16.h>
#include <stdint.h>

// Problem constants
#define B_    2
#define T_    2048
#define TOK_  512
#define POS_  512
#define H_    16
#define D_    64
#define KVH_  4

typedef __attribute__((ext_vector_type(16))) __bf16    bf16x16;
typedef __attribute__((ext_vector_type(8)))  float     f32x8;
typedef __attribute__((ext_vector_type(4)))  uint32_t  u32x4;

union Frag {
    bf16x16 m;
    u32x4   q[2];
};

__device__ __forceinline__ uint16_t f2bf(float f) {
    uint32_t u = __float_as_uint(f);
    u += 0x7fffu + ((u >> 16) & 1u);   // round-to-nearest-even
    return (uint16_t)(u >> 16);
}
__device__ __forceinline__ uint32_t pack2(float lo, float hi) {
    return (uint32_t)f2bf(lo) | ((uint32_t)f2bf(hi) << 16);
}

// ---------------------------------------------------------------------------
// f32 -> bf16 elementwise convert (2 elems/thread, packed 32-bit stores)
// ---------------------------------------------------------------------------
__global__ void k_convert_bf16(const float* __restrict__ src,
                               uint16_t* __restrict__ dst, int ntotal) {
    int i = (blockIdx.x * blockDim.x + threadIdx.x) * 2;
    if (i + 1 < ntotal) {
        *(uint32_t*)(dst + i) = pack2(src[i], src[i + 1]);
    }
}

// ---------------------------------------------------------------------------
// f32 [K,N] row-major  ->  bf16 [N,K] row-major (transposed weights so WMMA
// B-fragments become contiguous 16B loads per lane)
// ---------------------------------------------------------------------------
__global__ void k_transpose_bf16(const float* __restrict__ src,
                                 uint16_t* __restrict__ dst, int K, int N) {
    int idx = blockIdx.x * blockDim.x + threadIdx.x;
    if (idx < K * N) {
        int k = idx / N;
        int n = idx - k * N;
        dst[(size_t)n * K + k] = f2bf(src[idx]);
    }
}

// --------------------------- GEMM helpers ----------------------------------
__device__ __forceinline__ void load_a(Frag& f, const uint16_t* aptr, int kk,
                                       int h) {
    f.q[0] = *(const u32x4*)(aptr + kk + h * 8);
    f.q[1] = *(const u32x4*)(aptr + kk + 16 + h * 8);
}
__device__ __forceinline__ void load_b4(Frag (&f)[4],
                                        const uint16_t* const (&bptr)[4],
                                        int kk) {
#pragma unroll
    for (int j = 0; j < 4; ++j) {
        f[j].q[0] = *(const u32x4*)(bptr[j] + kk);
        f[j].q[1] = *(const u32x4*)(bptr[j] + kk + 16);
    }
}
__device__ __forceinline__ void mma4(f32x8 (&acc)[4], const Frag& a,
                                     const Frag (&b)[4]) {
#pragma unroll
    for (int j = 0; j < 4; ++j)
        acc[j] = __builtin_amdgcn_wmma_f32_16x16x32_bf16(
            false, a.m, false, b[j].m, (short)0, acc[j], false, false);
}

// ---------------------------------------------------------------------------
// Generic WMMA GEMM:  C[M,N] = A[M,Kd] * Bt[N,Kd]^T   (bf16 in, f32 accum)
// Block = 256 thr = 8 waves; wave tile 16(M) x 64(N); K step = 64 (2x32).
// Statically double-buffered (named buffers only -> no dynamic array index,
// no spill); launch_bounds(...,1) relaxes the VGPR budget. Requires
// Kdim % 64 == 0 (all call sites use 512 or 1024).
// mode 0: bf16 row-major store      (K projection)
// mode 1: bf16 store + per-head phase rotation (Q projection)
// mode 2: bf16 transposed store into Vt[B*256, T]     (V projection)
// mode 3: f32 row-major store       (final output GEMM)
// ---------------------------------------------------------------------------
__global__ __launch_bounds__(256, 1)
void k_wmma_gemm(const uint16_t* __restrict__ A, int lda,
                 const uint16_t* __restrict__ Bt, int ldb,
                 void* __restrict__ Cv, int ldc, int Kdim, int mode,
                 const float* __restrict__ angles) {
    const int tid  = threadIdx.x;
    const int lane = tid & 31;
    const int wave = tid >> 5;
    const int n    = lane & 15;   // column within 16-wide tile (C/B layout)
    const int h    = lane >> 4;   // lane-half selector
    const int m0   = blockIdx.x * 64 + (wave >> 1) * 16;
    const int n0   = blockIdx.y * 128 + (wave & 1) * 64;

    f32x8 acc[4];
#pragma unroll
    for (int j = 0; j < 4; ++j)
        for (int e = 0; e < 8; ++e) acc[j][e] = 0.0f;

    // A fragment: lanes 0-15 row m0+n (K=0..7,16..23); lanes 16-31 same rows,
    // K=8..15,24..31  -> two 16B loads per lane from row-major A.
    const uint16_t* aptr = A + (size_t)(m0 + n) * lda;
    const uint16_t* bptr[4];
#pragma unroll
    for (int j = 0; j < 4; ++j)
        bptr[j] = Bt + (size_t)(n0 + j * 16 + n) * ldb + h * 8;

    Frag afA, afB;
    Frag bA[4], bB[4];

    load_a(afA, aptr, 0, h);
    load_b4(bA, bptr, 0);

    for (int k0 = 0; k0 < Kdim; k0 += 64) {
        // Issue buffer-B loads (k0+32) before consuming buffer A, so the
        // pre-wmma wait only covers the older loads.
        load_a(afB, aptr, k0 + 32, h);
        load_b4(bB, bptr, k0 + 32);
        mma4(acc, afA, bA);
        if (k0 + 64 < Kdim) {
            load_a(afA, aptr, k0 + 64, h);
            load_b4(bA, bptr, k0 + 64);
        }
        mma4(acc, afB, bB);
    }

    if (mode == 1) {
        // Per-head phase rotation on (even,odd) head-dim pairs.
        // C layout: col = lane&15 -> pair partner sits in lane^1.
#pragma unroll
        for (int j = 0; j < 4; ++j) {
            int head = (n0 + j * 16) >> 6;     // 16-col tile fully inside a head
            float c = __cosf(angles[head]);
            float s = __sinf(angles[head]);
            bool even = ((n & 1) == 0);
#pragma unroll
            for (int r = 0; r < 8; ++r) {
                float v = acc[j][r];
                float p = __shfl_xor(v, 1, 32);
                acc[j][r] = even ? (v * c - p * s) : (p * s + v * c);
            }
        }
    }

    if (mode == 2) {
        // Transposed store: C(m=t within batch, col) -> Vt[b*256+col][t]
        uint16_t* C = (uint16_t*)Cv;
        int b     = m0 >> 11;           // m0 / T_
        int tbase = (m0 & (T_ - 1)) + h * 8;
#pragma unroll
        for (int j = 0; j < 4; ++j) {
            int col = n0 + j * 16 + n;
            u32x4 v;
            v[0] = pack2(acc[j][0], acc[j][1]);
            v[1] = pack2(acc[j][2], acc[j][3]);
            v[2] = pack2(acc[j][4], acc[j][5]);
            v[3] = pack2(acc[j][6], acc[j][7]);
            *(u32x4*)(C + (size_t)(b * (KVH_ * D_) + col) * T_ + tbase) = v;
        }
    } else if (mode == 3) {
        float* C = (float*)Cv;
#pragma unroll
        for (int j = 0; j < 4; ++j)
#pragma unroll
            for (int r = 0; r < 8; ++r)
                C[(size_t)(m0 + h * 8 + r) * ldc + n0 + j * 16 + n] = acc[j][r];
    } else {  // modes 0 and 1: bf16 row-major
        uint16_t* C = (uint16_t*)Cv;
#pragma unroll
        for (int j = 0; j < 4; ++j)
#pragma unroll
            for (int r = 0; r < 8; ++r)
                C[(size_t)(m0 + h * 8 + r) * ldc + n0 + j * 16 + n] =
                    f2bf(acc[j][r]);
    }
}

// ---------------------------------------------------------------------------
// Flash attention (causal, ALiBi, softmax-plus-one), GQA 16h -> 4kvh.
// Block = 128 thr = 4 waves; each wave owns one 16-row q-tile.
// Key blocks of 32: 4 QK^T wmma + online softmax + 4 AV wmma per block.
// V fragments are loaded up-front so they are in flight during the softmax
// VALU work and the LDS C->A re-layout of P (s_wait_dscnt 0).
// ---------------------------------------------------------------------------
__global__ __launch_bounds__(128, 1)
void k_flash_attn(const uint16_t* __restrict__ Qbf,   // [B,T,H*D]
                  const uint16_t* __restrict__ Kbf,   // [B,T,KVH*D]
                  const uint16_t* __restrict__ Vt,    // [B,KVH*D,T]
                  uint16_t* __restrict__ Obf,         // [B,T,H*D]
                  const float* __restrict__ log_slopes) {
    __shared__ uint16_t lds[4][16 * 32];

    const int tid  = threadIdx.x;
    const int lane = tid & 31;
    const int wave = tid >> 5;
    const int n    = lane & 15;
    const int h    = lane >> 4;
    const int bh   = blockIdx.y;
    const int b    = bh >> 4;
    const int head = bh & 15;
    const int kvh  = head >> 2;
    const int q0   = (blockIdx.x * 4 + wave) * 16;

    const float slope = __expf(log_slopes[head]);
    const float scale = 0.125f;   // 1/sqrt(64)

    // Q A-fragments, two d-chunks of 32 (row = q0 + (lane&15))
    const uint16_t* qptr =
        Qbf + (size_t)(b * T_ + q0 + n) * (H_ * D_) + head * D_;
    Frag qf0, qf1;
    qf0.q[0] = *(const u32x4*)(qptr + 0  + h * 8);
    qf0.q[1] = *(const u32x4*)(qptr + 16 + h * 8);
    qf1.q[0] = *(const u32x4*)(qptr + 32 + h * 8);
    qf1.q[1] = *(const u32x4*)(qptr + 48 + h * 8);

    f32x8 o[4];
#pragma unroll
    for (int j = 0; j < 4; ++j)
        for (int e = 0; e < 8; ++e) o[j][e] = 0.0f;
    float mrow[8], lrow[8];
#pragma unroll
    for (int r = 0; r < 8; ++r) { mrow[r] = -3.0e38f; lrow[r] = 0.0f; }

    const uint16_t* kbase = Kbf + (size_t)(b * T_) * (KVH_ * D_) + kvh * D_;
    const uint16_t* vbase = Vt + (size_t)(b * KVH_ * D_ + kvh * D_) * T_;
    uint16_t* myLds = lds[wave];

    for (int kb = 0; kb < q0 + 16; kb += 32) {
        // Issue V fragments early: independent of S, consumed last.
        Frag vf[4];
#pragma unroll
        for (int j = 0; j < 4; ++j) {
            const uint16_t* vp = vbase + (size_t)(j * 16 + n) * T_ + kb + h * 8;
            vf[j].q[0] = *(const u32x4*)(vp);
            vf[j].q[1] = *(const u32x4*)(vp + 16);
        }
        // Speculative prefetch of next key block's K rows (global_prefetch_b8)
        __builtin_prefetch(kbase + (size_t)(kb + 32 + n) * (KVH_ * D_), 0, 1);

        f32x8 S0, S1;
        for (int e = 0; e < 8; ++e) { S0[e] = 0.0f; S1[e] = 0.0f; }

        // S = Q * K^T  (B operand is K rows, d is the contraction dim)
        {
            const uint16_t* kp0 = kbase + (size_t)(kb + n) * (KVH_ * D_) + h * 8;
            const uint16_t* kp1 =
                kbase + (size_t)(kb + 16 + n) * (KVH_ * D_) + h * 8;
            Frag kf00, kf01, kf10, kf11;
            kf00.q[0] = *(const u32x4*)(kp0);
            kf00.q[1] = *(const u32x4*)(kp0 + 16);
            kf10.q[0] = *(const u32x4*)(kp1);
            kf10.q[1] = *(const u32x4*)(kp1 + 16);
            kf01.q[0] = *(const u32x4*)(kp0 + 32);
            kf01.q[1] = *(const u32x4*)(kp0 + 48);
            kf11.q[0] = *(const u32x4*)(kp1 + 32);
            kf11.q[1] = *(const u32x4*)(kp1 + 48);
            S0 = __builtin_amdgcn_wmma_f32_16x16x32_bf16(
                false, qf0.m, false, kf00.m, (short)0, S0, false, false);
            S1 = __builtin_amdgcn_wmma_f32_16x16x32_bf16(
                false, qf0.m, false, kf10.m, (short)0, S1, false, false);
            S0 = __builtin_amdgcn_wmma_f32_16x16x32_bf16(
                false, qf1.m, false, kf01.m, (short)0, S0, false, false);
            S1 = __builtin_amdgcn_wmma_f32_16x16x32_bf16(
                false, qf1.m, false, kf11.m, (short)0, S1, false, false);
        }

        // Online softmax (scale + ALiBi + causal mask; rows reduce over lanes)
        float p0[8], p1[8];
#pragma unroll
        for (int r = 0; r < 8; ++r) {
            int qpos = q0 + h * 8 + r;
            int c0 = kb + n, c1 = kb + 16 + n;
            float s0 = (c0 <= qpos)
                           ? (S0[r] * scale + slope * (float)(qpos - c0))
                           : -3.0e38f;
            float s1 = (c1 <= qpos)
                           ? (S1[r] * scale + slope * (float)(qpos - c1))
                           : -3.0e38f;
            float rm = fmaxf(s0, s1);
#pragma unroll
            for (int msk = 1; msk < 16; msk <<= 1)
                rm = fmaxf(rm, __shfl_xor(rm, msk, 32));
            float mnew = fmaxf(mrow[r], rm);
            float corr = __expf(mrow[r] - mnew);
            float e0 = __expf(s0 - mnew);
            float e1 = __expf(s1 - mnew);
            float rs = e0 + e1;
#pragma unroll
            for (int msk = 1; msk < 16; msk <<= 1)
                rs += __shfl_xor(rs, msk, 32);
            lrow[r] = lrow[r] * corr + rs;
            mrow[r] = mnew;
#pragma unroll
            for (int j = 0; j < 4; ++j) o[j][r] *= corr;
            p0[r] = e0;
            p1[r] = e1;
        }

        // C-layout P -> LDS -> A-layout fragment (16x32 bf16)
#pragma unroll
        for (int r = 0; r < 8; ++r) {
            myLds[(h * 8 + r) * 32 + n]      = f2bf(p0[r]);
            myLds[(h * 8 + r) * 32 + 16 + n] = f2bf(p1[r]);
        }
        asm volatile("s_wait_dscnt 0" ::: "memory");
        Frag pf;
        pf.q[0] = *(const u32x4*)(myLds + n * 32 + h * 8);
        pf.q[1] = *(const u32x4*)(myLds + n * 32 + 16 + h * 8);

        // O += P * V  (V fragments already resident)
#pragma unroll
        for (int j = 0; j < 4; ++j) {
            o[j] = __builtin_amdgcn_wmma_f32_16x16x32_bf16(
                false, pf.m, false, vf[j].m, (short)0, o[j], false, false);
        }
    }

    // Softmax-plus-one denominator, store bf16 attention output
    float inv[8];
#pragma unroll
    for (int r = 0; r < 8; ++r) inv[r] = 1.0f / (1.0f + lrow[r]);
#pragma unroll
    for (int j = 0; j < 4; ++j)
#pragma unroll
        for (int r = 0; r < 8; ++r)
            Obf[(size_t)(b * T_ + q0 + h * 8 + r) * (H_ * D_) + head * D_ +
                j * 16 + n] = f2bf(o[j][r] * inv[r]);
}

// ---------------------------------------------------------------------------
extern "C" void kernel_launch(void* const* d_in, const int* in_sizes, int n_in,
                              void* d_out, int out_size, void* d_ws,
                              size_t ws_size, hipStream_t stream) {
    (void)in_sizes; (void)n_in; (void)out_size; (void)ws_size;
    const float* x      = (const float*)d_in[0];
    const float* Wq     = (const float*)d_in[1];
    const float* Wk     = (const float*)d_in[2];
    const float* Wv     = (const float*)d_in[3];
    const float* Wo     = (const float*)d_in[4];
    const float* qphase = (const float*)d_in[5];
    const float* slopes = (const float*)d_in[6];
    float* out = (float*)d_out;

    // Workspace layout (bf16 elements)
    uint16_t* p = (uint16_t*)d_ws;
    uint16_t* xbf = p;  p += (size_t)B_ * T_ * 1024;          // x in bf16
    uint16_t* WqT = p;  p += (size_t)1024 * 512;              // Wq^T [1024,512]
    uint16_t* WkT = p;  p += (size_t)256 * 512;               // Wk^T [256,512]
    uint16_t* WvT = p;  p += (size_t)256 * 512;               // Wv^T [256,512]
    uint16_t* WoT = p;  p += (size_t)1024 * 1024;             // Wo^T [1024,1024]
    uint16_t* Qbf = p;  p += (size_t)B_ * T_ * (H_ * D_);     // [B,T,1024]
    uint16_t* Kbf = p;  p += (size_t)B_ * T_ * (KVH_ * D_);   // [B,T,256]
    uint16_t* Vtb = p;  p += (size_t)B_ * (KVH_ * D_) * T_;   // [B,256,T]
    uint16_t* Obf = p;  p += (size_t)B_ * T_ * (H_ * D_);     // [B,T,1024]

    const int M = B_ * T_;  // 4096, batch-flattened row dim

    // 1) Convert activations / transpose-convert weights to bf16
    {
        int nx = M * 1024;
        k_convert_bf16<<<dim3(nx / 512), dim3(256), 0, stream>>>(x, xbf, nx);
        k_transpose_bf16<<<dim3((512 * 1024) / 256), dim3(256), 0, stream>>>(
            Wq, WqT, 512, 1024);
        k_transpose_bf16<<<dim3((512 * 256) / 256), dim3(256), 0, stream>>>(
            Wk, WkT, 512, 256);
        k_transpose_bf16<<<dim3((512 * 256) / 256), dim3(256), 0, stream>>>(
            Wv, WvT, 512, 256);
        k_transpose_bf16<<<dim3((1024 * 1024) / 256), dim3(256), 0, stream>>>(
            Wo, WoT, 1024, 1024);
    }

    // 2) Projections (A column-offset selects x_pos / x_tok halves)
    // Q = x_pos @ Wq, rotated, bf16 [B,T,1024]
    k_wmma_gemm<<<dim3(M / 64, 1024 / 128), dim3(256), 0, stream>>>(
        xbf + TOK_, 1024, WqT, 512, Qbf, 1024, POS_, /*mode=*/1, qphase);
    // K = x_pos @ Wk, bf16 [B,T,256]
    k_wmma_gemm<<<dim3(M / 64, 256 / 128), dim3(256), 0, stream>>>(
        xbf + TOK_, 1024, WkT, 512, Kbf, 256, POS_, /*mode=*/0, nullptr);
    // V = x_tok @ Wv, transposed store -> Vt [B,256,T]
    k_wmma_gemm<<<dim3(M / 64, 256 / 128), dim3(256), 0, stream>>>(
        xbf, 1024, WvT, 512, Vtb, T_, TOK_, /*mode=*/2, nullptr);

    // 3) Flash attention: grid (q-tiles/4, B*H)
    k_flash_attn<<<dim3(T_ / 64, B_ * H_), dim3(128), 0, stream>>>(
        Qbf, Kbf, Vtb, Obf, slopes);

    // 4) out = O @ Wo, f32 [B,T,1024]
    k_wmma_gemm<<<dim3(M / 64, 1024 / 128), dim3(256), 0, stream>>>(
        Obf, 1024, WoT, 1024, out, 1024, /*Kdim=*/1024, /*mode=*/3, nullptr);
}
